// mamba_op_33724083208711
// MI455X (gfx1250) — matrix operand order repaired and checked by reading, not compiled, not run on hardware
//
#include <hip/hip_runtime.h>
#include <hip/hip_bf16.h>
#include <math.h>

#define D_MODEL   384
#define D_INNER   768
#define NHEADS    12
#define HEADDIM   64
#define DSTATE    64
#define CONV_DIM  896           // D_INNER + 2*DSTATE
#define D_IN_PROJ 1676          // 2*D_INNER + 2*DSTATE + NHEADS
#define PADN      1680          // 105 * 16  (padded N for in_proj GEMM)
#define LN_EPS    1e-5f
#define BATCH     8
#define HH        56
#define WW        56
#define LL        3136          // HH*WW
#define MROWS     (BATCH*LL)    // 25088

typedef __bf16 bf16_t;
typedef __attribute__((ext_vector_type(16))) __bf16 v16bf;
typedef __attribute__((ext_vector_type(8)))  __bf16 v8bf;
typedef __attribute__((ext_vector_type(4)))  __bf16 v4bf;
typedef __attribute__((ext_vector_type(8)))  float  v8f;

#if defined(__has_builtin)
# if __has_builtin(__builtin_amdgcn_tensor_load_to_lds)
#  define HAVE_TDM 1
# endif
#endif
#ifndef HAVE_TDM
# define HAVE_TDM 0
#endif

// ---- WMMA 16x16x32 bf16 fragment layout (CDNA5 ISA §7.12.2) ----
// A (16x32 MxK): lane&15 = M; elems 0..7 -> K = 8h..8h+7, elems 8..15 -> K = 16+8h.. (h = lane>=16)
// B (32x16 KxN): lane&15 = N; elems 0..7 -> K = 8h..8h+7, elems 8..15 -> K = 16+8h..
// C/D (16x16 f32): vgpr r -> M = r + 8h, N = lane&15
__device__ __forceinline__ v16bf cat8(v8bf lo, v8bf hi) {
    return __builtin_shufflevector(lo, hi, 0,1,2,3,4,5,6,7,8,9,10,11,12,13,14,15);
}
// A fragment from a bf16 row pointer (K contiguous): two b128 loads
__device__ __forceinline__ v16bf ld_a16(const bf16_t* row, int k0, int half) {
    return cat8(*reinterpret_cast<const v8bf*>(row + k0 + 8 * half),
                *reinterpret_cast<const v8bf*>(row + k0 + 16 + 8 * half));
}
// B fragment from a bf16 row pointer (K contiguous): two b128 loads
__device__ __forceinline__ v16bf ld_b16(const bf16_t* row, int k0, int half) {
    return cat8(*reinterpret_cast<const v8bf*>(row + k0 + 8 * half),
                *reinterpret_cast<const v8bf*>(row + k0 + 16 + 8 * half));
}
__device__ __forceinline__ float silu(float v) { return v / (1.0f + expf(-v)); }

#if HAVE_TDM
typedef __attribute__((ext_vector_type(4))) unsigned int v4u_t;
typedef __attribute__((ext_vector_type(8))) int v8i_t;
typedef __attribute__((ext_vector_type(4))) int v4i_t;
typedef __attribute__((address_space(3))) char lds_char_t;
__device__ __forceinline__ unsigned lds_off(void* p) {
    return (unsigned)(uintptr_t)(lds_char_t*)p;
}
// TDM 2D tile load (bf16 elements): D# per CDNA5 ISA
// This toolchain's builtin is the 6-arg form:
//   (uint32x4 g0, int32x8 g1, int32x4 g2, int32x4 g3, int32x8 extra, i32 cpol)
__device__ __forceinline__ void tdm_load_2d(unsigned lds_byte, const void* gptr,
                                            unsigned td0, unsigned td1,
                                            unsigned t0, unsigned t1, unsigned stride) {
    unsigned long long ga = (unsigned long long)(uintptr_t)gptr;
    v4u_t g0;
    g0[0] = 1u;                                        // count=1, user mode
    g0[1] = lds_byte;                                  // lds_addr
    g0[2] = (unsigned)ga;                              // global_addr[31:0]
    g0[3] = (unsigned)((ga >> 32) & 0x1FFFFFFu) | (2u << 30);  // ga[56:32] | type=2
    v8i_t g1;
    g1[0] = (int)(1u << 16);                           // wg_mask=0, data_size=1 (2B)
    g1[1] = (int)((td0 & 0xFFFFu) << 16);              // tensor_dim0 lo
    g1[2] = (int)(((td0 >> 16) & 0xFFFFu) | ((td1 & 0xFFFFu) << 16)); // dim0 hi | dim1 lo
    g1[3] = (int)(((td1 >> 16) & 0xFFFFu) | ((t0 & 0xFFFFu) << 16));  // dim1 hi | tile_dim0
    g1[4] = (int)(t1 & 0xFFFFu);                       // tile_dim1 (tile_dim2=0)
    g1[5] = (int)stride;                               // tensor_dim0_stride lo
    g1[6] = 0; g1[7] = 0;
    v4i_t z4 = {0, 0, 0, 0};
    v8i_t z8 = {0, 0, 0, 0, 0, 0, 0, 0};
    __builtin_amdgcn_tensor_load_to_lds(g0, g1, z4, z4, z8, 0);
}
#endif

// ---------------------------------------------------------------------------
// K0: fp32 -> bf16 bulk converter (x, in_proj_w, out_proj_w pre-pass)
// ---------------------------------------------------------------------------
__global__ __launch_bounds__(256)
void kcvt(const float* __restrict__ s, bf16_t* __restrict__ d, int n4) {
    int i = blockIdx.x * 256 + threadIdx.x;
    if (i >= n4) return;
    float4 f = reinterpret_cast<const float4*>(s)[i];
    v4bf v;
    v[0] = (bf16_t)f.x; v[1] = (bf16_t)f.y; v[2] = (bf16_t)f.z; v[3] = (bf16_t)f.w;
    reinterpret_cast<v4bf*>(d)[i] = v;
}

// ---------------------------------------------------------------------------
// K1: zxbcdt = x @ in_proj_w^T  (M=25088, N=1676->1680, K=384), bf16 operands
//     Weight tile (64 rows x 384) staged into LDS via TDM (async tensor DMA);
//     epilogue: softplus(dt + dt_bias) for columns [1664, 1676)
// ---------------------------------------------------------------------------
__global__ __launch_bounds__(128)
void k1_inproj(const bf16_t* __restrict__ xb, const bf16_t* __restrict__ wb,
               const float* __restrict__ dt_bias, float* __restrict__ zx) {
    __shared__ __align__(16) bf16_t Wl[64][D_MODEL];   // 48 KB
    const int tid  = threadIdx.x;
    const int lane = tid & 31;
    const int wave = tid >> 5;
    const int half = lane >> 4;
    const int lid  = lane & 15;
    const int m0 = (blockIdx.y * 4 + wave) * 16;   // 1568 m-tiles
    const int n0 = blockIdx.x * 64;                // 27 n-groups (covers 105 tiles)

#if HAVE_TDM
    if (wave == 0) {
        unsigned rows = (unsigned)(D_IN_PROJ - n0); if (rows > 64u) rows = 64u;
        tdm_load_2d(lds_off(&Wl[0][0]), wb + (size_t)n0 * D_MODEL,
                    /*td0=*/D_MODEL, /*td1=*/rows, /*t0=*/D_MODEL, /*t1=*/64u,
                    /*stride=*/D_MODEL);
        __builtin_amdgcn_s_wait_tensorcnt(0);
    }
#else
    for (int i = tid * 4; i < 64 * D_MODEL; i += 128 * 4) {
        int row = i / D_MODEL, col = i % D_MODEL;
        int nr = n0 + row; if (nr > D_IN_PROJ - 1) nr = D_IN_PROJ - 1;
        *reinterpret_cast<v4bf*>(&Wl[row][col]) =
            *reinterpret_cast<const v4bf*>(&wb[(size_t)nr * D_MODEL + col]);
    }
#endif
    __syncthreads();

    const bf16_t* xrow = xb + (size_t)(m0 + lid) * D_MODEL;
    v8f acc[4] = {};
    for (int k0 = 0; k0 < D_MODEL; k0 += 32) {
        v16bf a = ld_a16(xrow, k0, half);
#pragma unroll
        for (int t = 0; t < 4; ++t) {
            v16bf bfr = ld_b16(&Wl[t * 16 + lid][0], k0, half);
            acc[t] = __builtin_amdgcn_wmma_f32_16x16x32_bf16(false, a, false, bfr,
                                                             (short)0, acc[t], false, false);
        }
    }
#pragma unroll
    for (int t = 0; t < 4; ++t) {
#pragma unroll
        for (int r = 0; r < 8; ++r) {
            int m = m0 + r + 8 * half;
            int n = n0 + t * 16 + lid;
            if (n < PADN) {
                float v = acc[t][r];
                if (n >= D_INNER + CONV_DIM && n < D_IN_PROJ) {
                    float tt = v + dt_bias[n - (D_INNER + CONV_DIM)];
                    v = (tt > 20.0f) ? tt : log1pf(expf(tt));     // softplus
                }
                zx[(size_t)m * PADN + n] = v;
            }
        }
    }
}

// ---------------------------------------------------------------------------
// K2: depthwise 3x3 conv (SAME) + bias + SiLU; writes fp32 co, plus bf16
//     transposed Bm copy (BmT[b][s][L]) and bf16 Cm copy (cobc[row][64])
// ---------------------------------------------------------------------------
__global__ __launch_bounds__(128)
void k2_conv(const float* __restrict__ zx, const float* __restrict__ cw,
             const float* __restrict__ cb, float* __restrict__ co,
             bf16_t* __restrict__ BmT, bf16_t* __restrict__ cobc) {
    const int c  = blockIdx.x * 128 + threadIdx.x;   // 7*128 = 896 exact
    const int hw = blockIdx.y;                        // 0..3135
    const int b  = blockIdx.z;
    const int h = hw / WW, w = hw % WW;
    float s = cb[c];
#pragma unroll
    for (int dy = -1; dy <= 1; ++dy) {
#pragma unroll
        for (int dx = -1; dx <= 1; ++dx) {
            int hh = h + dy, ww = w + dx;
            if (hh >= 0 && hh < HH && ww >= 0 && ww < WW) {
                s += zx[(size_t)(b * LL + hh * WW + ww) * PADN + D_INNER + c]
                     * cw[c * 9 + (dy + 1) * 3 + (dx + 1)];
            }
        }
    }
    float v = silu(s);
    co[(size_t)(b * LL + hw) * CONV_DIM + c] = v;
    if (c >= D_INNER + DSTATE) {                      // Cm: row-major bf16
        cobc[(size_t)(b * LL + hw) * DSTATE + (c - D_INNER - DSTATE)] = (bf16_t)v;
    } else if (c >= D_INNER) {                        // Bm: transposed bf16
        BmT[((size_t)b * DSTATE + (c - D_INNER)) * LL + hw] = (bf16_t)v;
    }
}

// ---------------------------------------------------------------------------
// K3ab: fused dx = dt*A*xs + 3x3 count-normalized avg pool; output TRANSPOSED
//       bf16: dxpT[bh][d][L]  (staged through LDS so global stores are wide)
//       block = (b, head, 64-l chunk); 128 threads
// ---------------------------------------------------------------------------
__global__ __launch_bounds__(128)
void k3ab_dxpool(const float* __restrict__ zx, const float* __restrict__ co,
                 const float* __restrict__ A_log, bf16_t* __restrict__ dxpT) {
    __shared__ __align__(16) bf16_t Tl[HEADDIM][64];   // [d][l_local] 8 KB
    const int head = blockIdx.y, b = blockIdx.z;
    const int l0 = blockIdx.x * 64;
    const int tid = threadIdx.x;
    const int d = tid & 63;
    const float Ahd = -expf(A_log[head * HEADDIM + d]);

#pragma unroll
    for (int pass = 0; pass < 32; ++pass) {
        int l_loc = (tid >> 6) + 2 * pass;             // 0..63
        int l = l0 + l_loc;
        int h = l / WW, w = l % WW;
        float sum = 0.0f; int cnt = 0;
        for (int dy = -1; dy <= 1; ++dy) {
            for (int dx = -1; dx <= 1; ++dx) {
                int hh = h + dy, ww = w + dx;
                if (hh >= 0 && hh < HH && ww >= 0 && ww < WW) {
                    int ln = hh * WW + ww;
                    float dtv = zx[(size_t)(b * LL + ln) * PADN + (D_INNER + CONV_DIM) + head];
                    float xsv = co[(size_t)(b * LL + ln) * CONV_DIM + head * HEADDIM + d];
                    sum += dtv * xsv; ++cnt;
                }
            }
        }
        Tl[d][l_loc] = (bf16_t)(Ahd * sum / (float)cnt);
    }
    __syncthreads();
    // write out: 2 threads per d-row, 32 bf16 (64B) each, vectorized
    const int dr = tid >> 1, hseg = (tid & 1) * 32;
    bf16_t* dst = dxpT + ((size_t)(b * NHEADS + head) * HEADDIM + dr) * LL + l0 + hseg;
#pragma unroll
    for (int q = 0; q < 4; ++q)
        *reinterpret_cast<v8bf*>(dst + q * 8) =
            *reinterpret_cast<const v8bf*>(&Tl[dr][hseg + q * 8]);
}

// ---------------------------------------------------------------------------
// K3c: h_state[bh,s,d] = sum_l BmT[b,s,l] * dxpT[bh,d,l]   (64x64, K=3136)
//      both operands pre-transposed bf16 -> pure register GEMM, no LDS
//      1 block (16 waves) per (b,h); each wave one 16x16 tile
// ---------------------------------------------------------------------------
__global__ __launch_bounds__(512)
void k3c_hstate(const bf16_t* __restrict__ BmT, const bf16_t* __restrict__ dxpT,
                float* __restrict__ hs) {
    const int bh = blockIdx.x, b = bh / NHEADS;
    const int tid = threadIdx.x, lane = tid & 31, wave = tid >> 5;
    const int half = lane >> 4, lid = lane & 15;
    const int s0 = (wave >> 2) * 16, d0 = (wave & 3) * 16;

    const bf16_t* arow = BmT + ((size_t)b * DSTATE + s0 + lid) * LL;
    const bf16_t* brow = dxpT + ((size_t)bh * HEADDIM + d0 + lid) * LL;
    v8f acc = {};
    for (int k0 = 0; k0 < LL; k0 += 32) {  // 98 iterations
        v16bf a   = ld_a16(arow, k0, half);
        v16bf bfr = ld_b16(brow, k0, half);
        acc = __builtin_amdgcn_wmma_f32_16x16x32_bf16(false, a, false, bfr,
                                                      (short)0, acc, false, false);
    }
#pragma unroll
    for (int r = 0; r < 8; ++r) {
        int s = s0 + r + 8 * half, d = d0 + lid;
        hs[(size_t)bh * (DSTATE * HEADDIM) + s * HEADDIM + d] = acc[r];
    }
}

// ---------------------------------------------------------------------------
// K3d: Ch[bh,l,d] = sum_s Cm[b,l,s] * h_state[bh,s,d]   (M=3136, N=64, K=64)
//      Cm is bf16 K-contiguous (cobc); h_state LDS-transposed bf16
// ---------------------------------------------------------------------------
__global__ __launch_bounds__(128)
void k3d_ch(const bf16_t* __restrict__ cobc, const float* __restrict__ hs,
            float* __restrict__ ch) {
    __shared__ __align__(16) bf16_t Hsl[HEADDIM][DSTATE];   // [d][s] 8 KB
    const int bh = blockIdx.y, b = bh / NHEADS;
    const int tid  = threadIdx.x;
    const int lane = tid & 31;
    const int wave = tid >> 5;             // 0..3 -> d tile
    const int half = lane >> 4, lid = lane & 15;
    const int m0 = blockIdx.x * 16;        // 196 m-tiles
    const int d0 = wave * 16;

#pragma unroll
    for (int pass = 0; pass < 8; ++pass) {
        int i4 = (tid + pass * 128) * 4;
        float4 f = *reinterpret_cast<const float4*>(&hs[(size_t)bh * 4096 + i4]);
        int s = i4 >> 6, d = i4 & 63;
        Hsl[d + 0][s] = (bf16_t)f.x; Hsl[d + 1][s] = (bf16_t)f.y;
        Hsl[d + 2][s] = (bf16_t)f.z; Hsl[d + 3][s] = (bf16_t)f.w;
    }
    __syncthreads();

    const bf16_t* arow = cobc + (size_t)(b * LL + m0 + lid) * DSTATE;
    v8f acc = {};
#pragma unroll
    for (int k0 = 0; k0 < DSTATE; k0 += 32) {
        v16bf a   = ld_a16(arow, k0, half);
        v16bf bfr = ld_b16(&Hsl[d0 + lid][0], k0, half);
        acc = __builtin_amdgcn_wmma_f32_16x16x32_bf16(false, a, false, bfr,
                                                      (short)0, acc, false, false);
    }
#pragma unroll
    for (int r = 0; r < 8; ++r) {
        int l = m0 + r + 8 * half, d = d0 + lid;
        ch[((size_t)bh * LL + l) * HEADDIM + d] = acc[r];
    }
}

// ---------------------------------------------------------------------------
// K4: g = (Ch + xs*D) * silu(z); layernorm(768) -> bf16 gbuf
// ---------------------------------------------------------------------------
__global__ __launch_bounds__(256)
void k4_gate_ln(const float* __restrict__ zx, const float* __restrict__ co,
                const float* __restrict__ ch, const float* __restrict__ Dp,
                const float* __restrict__ nw, const float* __restrict__ nb,
                bf16_t* __restrict__ g) {
    const int row = blockIdx.x;            // b*LL + l
    const int b = row / LL, l = row % LL;
    const int tid = threadIdx.x;
    __shared__ float sbuf[256];

    float gv[3];
    float s1 = 0.0f;
#pragma unroll
    for (int j = 0; j < 3; ++j) {
        int c = tid + 256 * j;
        int head = c >> 6, d = c & 63;
        float y = ch[((size_t)(b * NHEADS + head) * LL + l) * HEADDIM + d]
                  + co[(size_t)row * CONV_DIM + c] * Dp[head];
        float z = zx[(size_t)row * PADN + c];
        gv[j] = y * silu(z);
        s1 += gv[j];
    }
    sbuf[tid] = s1; __syncthreads();
    for (int off = 128; off > 0; off >>= 1) {
        if (tid < off) sbuf[tid] += sbuf[tid + off];
        __syncthreads();
    }
    float mu = sbuf[0] * (1.0f / D_INNER);
    __syncthreads();
    float s2 = 0.0f;
#pragma unroll
    for (int j = 0; j < 3; ++j) { float dd = gv[j] - mu; s2 += dd * dd; }
    sbuf[tid] = s2; __syncthreads();
    for (int off = 128; off > 0; off >>= 1) {
        if (tid < off) sbuf[tid] += sbuf[tid + off];
        __syncthreads();
    }
    float rstd = rsqrtf(sbuf[0] * (1.0f / D_INNER) + LN_EPS);
#pragma unroll
    for (int j = 0; j < 3; ++j) {
        int c = tid + 256 * j;
        g[(size_t)row * D_INNER + c] = (bf16_t)((gv[j] - mu) * rstd * nw[c] + nb[c]);
    }
}

// ---------------------------------------------------------------------------
// K5: out = g @ out_proj_w^T   (M=25088, N=384, K=768), bf16 operands
// ---------------------------------------------------------------------------
__global__ __launch_bounds__(128)
void k5_outproj(const bf16_t* __restrict__ g, const bf16_t* __restrict__ w,
                float* __restrict__ out) {
    const int lane = threadIdx.x & 31;
    const int wave = threadIdx.x >> 5;
    const int half = lane >> 4, lid = lane & 15;
    const int m0 = (blockIdx.y * 4 + wave) * 16;   // 1568 m-tiles
    const int n0 = blockIdx.x * 64;                // 6 n-groups (24 tiles exact)

    const bf16_t* grow = g + (size_t)(m0 + lid) * D_INNER;
    const bf16_t* wrow[4];
#pragma unroll
    for (int t = 0; t < 4; ++t)
        wrow[t] = w + (size_t)(n0 + t * 16 + lid) * D_INNER;

    v8f acc[4] = {};
    for (int k0 = 0; k0 < D_INNER; k0 += 32) {     // 24 iterations
        v16bf a = ld_a16(grow, k0, half);
#pragma unroll
        for (int t = 0; t < 4; ++t) {
            v16bf bfr = ld_b16(wrow[t], k0, half);
            acc[t] = __builtin_amdgcn_wmma_f32_16x16x32_bf16(false, a, false, bfr,
                                                             (short)0, acc[t], false, false);
        }
    }
#pragma unroll
    for (int t = 0; t < 4; ++t) {
#pragma unroll
        for (int r = 0; r < 8; ++r) {
            int m = m0 + r + 8 * half;
            out[(size_t)m * D_MODEL + n0 + t * 16 + lid] = acc[t][r];
        }
    }
}

// ---------------------------------------------------------------------------
extern "C" void kernel_launch(void* const* d_in, const int* in_sizes, int n_in,
                              void* d_out, int out_size, void* d_ws, size_t ws_size,
                              hipStream_t stream) {
    const float* x       = (const float*)d_in[0];
    const float* ipw     = (const float*)d_in[1];
    const float* cw      = (const float*)d_in[2];
    const float* cb      = (const float*)d_in[3];
    const float* dt_bias = (const float*)d_in[4];
    const float* A_log   = (const float*)d_in[5];
    const float* Dp      = (const float*)d_in[6];
    const float* nw      = (const float*)d_in[7];
    const float* nb      = (const float*)d_in[8];
    const float* opw     = (const float*)d_in[9];
    float* out = (float*)d_out;

    // workspace carve-up
    char* p = (char*)d_ws;
    auto carve = [&](size_t bytes) { char* r = p; p += (bytes + 255) & ~(size_t)255; return r; };
    float*  zx   = (float*) carve((size_t)MROWS * PADN * 4);
    float*  co   = (float*) carve((size_t)MROWS * CONV_DIM * 4);
    float*  chb  = (float*) carve((size_t)MROWS * D_INNER * 4);
    float*  hs   = (float*) carve((size_t)96 * 4096 * 4);
    bf16_t* xb   = (bf16_t*)carve((size_t)MROWS * D_MODEL * 2);
    bf16_t* ipwb = (bf16_t*)carve((size_t)D_IN_PROJ * D_MODEL * 2);
    bf16_t* opwb = (bf16_t*)carve((size_t)D_MODEL * D_INNER * 2);
    bf16_t* BmT  = (bf16_t*)carve((size_t)BATCH * DSTATE * LL * 2);
    bf16_t* cobc = (bf16_t*)carve((size_t)MROWS * DSTATE * 2);
    bf16_t* dxpT = (bf16_t*)carve((size_t)MROWS * D_INNER * 2);
    bf16_t* gb   = (bf16_t*)carve((size_t)MROWS * D_INNER * 2);

    // pre-pass: convert x / weights to bf16 once
    {
        int n4 = MROWS * D_MODEL / 4;
        kcvt<<<(n4 + 255) / 256, 256, 0, stream>>>(x, xb, n4);
        n4 = D_IN_PROJ * D_MODEL / 4;
        kcvt<<<(n4 + 255) / 256, 256, 0, stream>>>(ipw, ipwb, n4);
        n4 = D_MODEL * D_INNER / 4;
        kcvt<<<(n4 + 255) / 256, 256, 0, stream>>>(opw, opwb, n4);
    }

    dim3 g1(27, MROWS / 64);
    k1_inproj<<<g1, 128, 0, stream>>>(xb, ipwb, dt_bias, zx);

    dim3 g2(CONV_DIM / 128, LL, BATCH);
    k2_conv<<<g2, 128, 0, stream>>>(zx, cw, cb, co, BmT, cobc);

    dim3 g3(LL / 64, NHEADS, BATCH);
    k3ab_dxpool<<<g3, 128, 0, stream>>>(zx, co, A_log, dxpT);

    k3c_hstate<<<BATCH * NHEADS, 512, 0, stream>>>(BmT, dxpT, hs);

    dim3 g3d(LL / 16, BATCH * NHEADS);
    k3d_ch<<<g3d, 128, 0, stream>>>(cobc, hs, chb);

    k4_gate_ln<<<MROWS, 256, 0, stream>>>(zx, co, chb, Dp, nw, nb, gb);

    dim3 g5(D_MODEL / 64, MROWS / 64);
    k5_outproj<<<g5, 128, 0, stream>>>(gb, opwb, out);
}
